// Attention_90589450207742
// MI455X (gfx1250) — matrix operand order repaired and checked by reading, not compiled, not run on hardware
//
#include <hip/hip_runtime.h>
#include <math.h>

typedef __attribute__((ext_vector_type(16))) __bf16 v16bf;
typedef __attribute__((ext_vector_type(8)))  float  v8f;

#define HEADS 4
#define DH    128
#define NPOS  4096   // H*W
#define CIN   512
#define RELR  128    // 127 padded to 128
// q scale = DH^-0.5 * log2(e): moves softmax into exp2 domain (exact rewrite)
#define QSCALE (0.08838834764831845f * 1.4426950408889634f)

// ---------- helpers ----------
__device__ __forceinline__ unsigned short f2bf(float x) {
    return __builtin_bit_cast(unsigned short, (__bf16)x);   // single v_cvt
}

union FragU { uint4 u[2]; v16bf v; };

// A-fragment (16x32 bf16, row-major source): lane (idx,half) holds row `row`,
// K = {c0+8*half .. +7} and {c0+16+8*half .. +7}   (ISA 7.12.2 16-bit A layout)
__device__ __forceinline__ v16bf load_a(const unsigned short* base, int row, int lda,
                                        int c0, int half) {
    const unsigned short* r = base + (size_t)row * lda + c0;
    FragU f;
    f.u[0] = *(const uint4*)(r + 8 * half);
    f.u[1] = *(const uint4*)(r + 16 + 8 * half);
    return f.v;
}
// B-fragment (32x16 bf16): lane holds column `row` (storage row = N index),
// K = {c0+8*half .. +7} and {c0+16+8*half .. +7}  (same K order as load_a)
__device__ __forceinline__ v16bf load_b(const unsigned short* base, int row, int lda,
                                        int c0, int half) {
    const unsigned short* r = base + (size_t)row * lda + c0;
    FragU f;
    f.u[0] = *(const uint4*)(r + 8 * half);
    f.u[1] = *(const uint4*)(r + 16 + 8 * half);
    return f.v;
}

__device__ __forceinline__ v8f wmma_bf16(v16bf a, v16bf b, v8f c) {
    return __builtin_amdgcn_wmma_f32_16x16x32_bf16(false, a, false, b, (short)0, c,
                                                   false, false);
}

// wave-local LDS fence: same-wave DS ops are in-order; this stops compiler
// reordering and drains DScnt before the cross-lane re-striping read.
__device__ __forceinline__ void lds_fence() {
    asm volatile("s_wait_dscnt 0x0" ::: "memory");
}

// ---------- conversion kernels ----------
__global__ __launch_bounds__(256) void k_conv_fmap_t(const float* __restrict__ src,
                                                     unsigned short* __restrict__ dst) {
    int i = blockIdx.x * blockDim.x + threadIdx.x;   // exactly CIN*NPOS threads
    int c = i >> 12, p = i & 4095;
    dst[(size_t)p * CIN + c] = f2bf(src[i]);         // fmap[c][p] -> fmap_t[p][c]
}

__global__ __launch_bounds__(256) void k_conv(const float* __restrict__ src,
                                              unsigned short* __restrict__ dst, int n) {
    int i = blockIdx.x * blockDim.x + threadIdx.x;
    if (i < n) dst[i] = f2bf(src[i]);
}

__global__ __launch_bounds__(256) void k_conv_rel(const float* __restrict__ src,
                                                  unsigned short* __restrict__ dst) {
    int i = blockIdx.x * blockDim.x + threadIdx.x;   // 128*128 threads
    dst[i] = (i < 127 * 128) ? f2bf(src[i]) : (unsigned short)0;  // zero pad row 127
}

// ---------- QKV projection: D(o,p) = sum_c w[o][c] * fmap_t[p][c] ----------
__global__ __launch_bounds__(128) void k_qkv(const unsigned short* __restrict__ wb,
                                             const unsigned short* __restrict__ xb,
                                             unsigned short* __restrict__ qb,
                                             unsigned short* __restrict__ kb,
                                             unsigned short* __restrict__ vtb) {
    int wave = threadIdx.x >> 5, lane = threadIdx.x & 31;
    int idx = lane & 15, half = lane >> 4;
    int o0 = (blockIdx.x * 4 + wave) * 16;   // 0..1535 in steps of 16
    int p0 = blockIdx.y * 16;                // 0..4095 in steps of 16

    v8f acc = {};
    for (int kc = 0; kc < CIN; kc += 32) {
        v16bf a = load_a(wb, o0 + idx, CIN, kc, half);
        v16bf b = load_b(xb, p0 + idx, CIN, kc, half);
        acc = wmma_bf16(a, b, acc);
    }
    // C tile: lane column p = p0+idx, rows o = o0 + 8*half + v
    int p = p0 + idx;
    int dbase = (o0 + 8 * half) & 127;
    if (o0 < 512) {                   // q: store log2-scaled bf16 [head][p][d]
        int head = o0 >> 7;
        union { unsigned short h[8]; uint4 u; } pk;
        #pragma unroll
        for (int v = 0; v < 8; ++v) pk.h[v] = f2bf(acc[v] * QSCALE);
        *(uint4*)(qb + (size_t)(head * NPOS + p) * DH + dbase) = pk.u;
    } else if (o0 < 1024) {           // k: bf16 [head][p][d]
        int head = (o0 - 512) >> 7;
        union { unsigned short h[8]; uint4 u; } pk;
        #pragma unroll
        for (int v = 0; v < 8; ++v) pk.h[v] = f2bf(acc[v]);
        *(uint4*)(kb + (size_t)(head * NPOS + p) * DH + dbase) = pk.u;
    } else {                          // v: transposed bf16 [head][d][p]
        int head = (o0 - 1024) >> 7;
        #pragma unroll
        for (int v = 0; v < 8; ++v)
            vtb[(size_t)(head * DH + dbase + v) * NPOS + p] = f2bf(acc[v]);
    }
}

// ---------- relative-logit GEMM: R[h][p][r] = q[h][p][:] . rel[r][:] ----------
__global__ __launch_bounds__(128) void k_rel(const unsigned short* __restrict__ qb,
                                             const unsigned short* __restrict__ relb,
                                             float* __restrict__ outR) {
    int wave = threadIdx.x >> 5, lane = threadIdx.x & 31;
    int idx = lane & 15, half = lane >> 4;
    int r0 = (blockIdx.x * 4 + wave) * 16;   // 0..127
    int p0 = blockIdx.y * 16;
    int head = blockIdx.z;
    const unsigned short* qh = qb + (size_t)head * NPOS * DH;

    v8f acc = {};
    for (int kc = 0; kc < DH; kc += 32) {
        v16bf a = load_a(qh, p0 + idx, DH, kc, half);
        v16bf b = load_b(relb, r0 + idx, DH, kc, half);
        acc = wmma_bf16(a, b, acc);
    }
    int r = r0 + idx;
    int pbase = p0 + 8 * half;
    float* dst = outR + (size_t)head * NPOS * RELR;
    #pragma unroll
    for (int v = 0; v < 8; ++v)
        dst[(size_t)(pbase + v) * RELR + r] = acc[v];
}

// ---------- flash attention with relative bias (1 wave / workgroup) ----------
__global__ __launch_bounds__(32) void k_attn(const unsigned short* __restrict__ qb,
                                             const unsigned short* __restrict__ kb,
                                             const unsigned short* __restrict__ vtb,
                                             const float* __restrict__ RW,
                                             const float* __restrict__ RH,
                                             float* __restrict__ out) {
    __shared__ __align__(16) unsigned short pl[4 * 256];  // 4 P subtiles, bf16
    int lane = threadIdx.x;
    int idx = lane & 15, half = lane >> 4;
    int head = blockIdx.y;
    int q0 = blockIdx.x * 16;

    const unsigned short* qh = qb  + (size_t)head * NPOS * DH;
    const unsigned short* kh = kb  + (size_t)head * NPOS * DH;
    const unsigned short* vh = vtb + (size_t)head * DH * NPOS;
    const float* RWh = RW + (size_t)head * NPOS * RELR;
    const float* RHh = RH + (size_t)head * NPOS * RELR;

    int pb = q0 + 8 * half;            // first of this lane's 8 query rows

    // RW bias is independent of the key block (j = 16t+idx always): hoist to regs.
    float rwv[4][8];
    #pragma unroll
    for (int t = 0; t < 4; ++t)
        #pragma unroll
        for (int v = 0; v < 8; ++v) {
            int p = pb + v;
            rwv[t][v] = RWh[(size_t)p * RELR + (16 * t + idx - (p & 63) + 63)];
        }

    v8f z = {};
    v8f o[8];
    #pragma unroll
    for (int t = 0; t < 8; ++t) o[t] = z;
    float mrun[8], lrun[8];
    #pragma unroll
    for (int v = 0; v < 8; ++v) { mrun[v] = -3.0e38f; lrun[v] = 0.0f; }

    for (int kb0 = 0; kb0 < NPOS; kb0 += 64) {
        // --- S = Q.K^T for four 16-key subtiles (log2 domain) ---
        v8f s[4];
        #pragma unroll
        for (int t = 0; t < 4; ++t) s[t] = z;
        #pragma unroll
        for (int c = 0; c < 4; ++c) {
            v16bf a = load_a(qh, q0 + idx, DH, 32 * c, half);
            #pragma unroll
            for (int t = 0; t < 4; ++t) {
                v16bf b = load_b(kh, kb0 + 16 * t + idx, DH, 32 * c, half);
                s[t] = wmma_bf16(a, b, s[t]);
            }
        }
        // --- bias: RH constant per 64-key block (i = kb0>>6), RW from regs ---
        int iblk = kb0 >> 6;
        float rh[8];
        #pragma unroll
        for (int v = 0; v < 8; ++v) {
            int p = pb + v;
            rh[v] = RHh[(size_t)p * RELR + (iblk - (p >> 6) + 63)];
        }
        #pragma unroll
        for (int t = 0; t < 4; ++t)
            #pragma unroll
            for (int v = 0; v < 8; ++v) s[t][v] += rwv[t][v] + rh[v];
        // --- online softmax over 64 keys (row groups = 16-lane halves) ---
        float scalef[8];
        #pragma unroll
        for (int v = 0; v < 8; ++v) {
            float cm = fmaxf(fmaxf(s[0][v], s[1][v]), fmaxf(s[2][v], s[3][v]));
            cm = fmaxf(cm, __shfl_xor(cm, 1));
            cm = fmaxf(cm, __shfl_xor(cm, 2));
            cm = fmaxf(cm, __shfl_xor(cm, 4));
            cm = fmaxf(cm, __shfl_xor(cm, 8));
            float mn = fmaxf(mrun[v], cm);
            scalef[v] = exp2f(mrun[v] - mn);
            mrun[v] = mn;
        }
        #pragma unroll
        for (int v = 0; v < 8; ++v) {
            float e0 = exp2f(s[0][v] - mrun[v]);
            float e1 = exp2f(s[1][v] - mrun[v]);
            float e2 = exp2f(s[2][v] - mrun[v]);
            float e3 = exp2f(s[3][v] - mrun[v]);
            s[0][v] = e0; s[1][v] = e1; s[2][v] = e2; s[3][v] = e3;
            float r = (e0 + e1) + (e2 + e3);
            r += __shfl_xor(r, 1); r += __shfl_xor(r, 2);
            r += __shfl_xor(r, 4); r += __shfl_xor(r, 8);
            lrun[v] = lrun[v] * scalef[v] + r;
        }
        // --- re-stripe P (C-layout) into A-layout via LDS, as bf16 ---
        #pragma unroll
        for (int t = 0; t < 4; ++t)
            #pragma unroll
            for (int v = 0; v < 8; ++v)
                pl[t * 256 + (v + 8 * half) * 16 + idx] = f2bf(s[t][v]);
        lds_fence();
        FragU ap0, ap1;
        ap0.u[0] = *(const uint4*)(pl + idx * 16 + 8 * half);          // keys 0-31
        ap0.u[1] = *(const uint4*)(pl + 256 + idx * 16 + 8 * half);
        ap1.u[0] = *(const uint4*)(pl + 512 + idx * 16 + 8 * half);    // keys 32-63
        ap1.u[1] = *(const uint4*)(pl + 768 + idx * 16 + 8 * half);
        lds_fence();
        // --- O = O*scale + P.V  (V^T gives contiguous key runs per lane) ---
        #pragma unroll
        for (int dt = 0; dt < 8; ++dt) {
            #pragma unroll
            for (int v = 0; v < 8; ++v) o[dt][v] *= scalef[v];
            v16bf bv0 = load_b(vh, dt * 16 + idx, NPOS, kb0, half);
            v16bf bv1 = load_b(vh, dt * 16 + idx, NPOS, kb0 + 32, half);
            o[dt] = wmma_bf16(ap0.v, bv0, o[dt]);
            o[dt] = wmma_bf16(ap1.v, bv1, o[dt]);
        }
    }
    // --- normalize and store: out[(head*DH+d)][p], contiguous in p ---
    float inv[8];
    #pragma unroll
    for (int v = 0; v < 8; ++v) inv[v] = 1.0f / lrun[v];
    #pragma unroll
    for (int dt = 0; dt < 8; ++dt) {
        int d = dt * 16 + idx;
        float* op = out + (size_t)(head * DH + d) * NPOS + pb;
        float4 a4, b4;
        a4.x = o[dt][0] * inv[0]; a4.y = o[dt][1] * inv[1];
        a4.z = o[dt][2] * inv[2]; a4.w = o[dt][3] * inv[3];
        b4.x = o[dt][4] * inv[4]; b4.y = o[dt][5] * inv[5];
        b4.z = o[dt][6] * inv[6]; b4.w = o[dt][7] * inv[7];
        *(float4*)(op)     = a4;
        *(float4*)(op + 4) = b4;
    }
}

// ---------- host ----------
extern "C" void kernel_launch(void* const* d_in, const int* in_sizes, int n_in,
                              void* d_out, int out_size, void* d_ws, size_t ws_size,
                              hipStream_t stream) {
    const float* fmap = (const float*)d_in[0];   // (512,64,64)
    const float* wqkv = (const float*)d_in[1];   // (1536,512)
    const float* relh = (const float*)d_in[2];   // (127,128)
    const float* relw = (const float*)d_in[3];   // (127,128)
    float* out = (float*)d_out;                  // (512,64,64)

    char* ws = (char*)d_ws;
    size_t off = 0;
    auto nxt = [&](size_t bytes) { size_t o = off; off += (bytes + 255) & ~(size_t)255; return o; };
    unsigned short* fmap_t  = (unsigned short*)(ws + nxt((size_t)NPOS * CIN * 2));
    unsigned short* w_bf    = (unsigned short*)(ws + nxt((size_t)1536 * CIN * 2));
    unsigned short* relw_bf = (unsigned short*)(ws + nxt((size_t)RELR * DH * 2));
    unsigned short* relh_bf = (unsigned short*)(ws + nxt((size_t)RELR * DH * 2));
    unsigned short* q_bf    = (unsigned short*)(ws + nxt((size_t)HEADS * NPOS * DH * 2));
    unsigned short* k_bf    = (unsigned short*)(ws + nxt((size_t)HEADS * NPOS * DH * 2));
    unsigned short* vt_bf   = (unsigned short*)(ws + nxt((size_t)HEADS * NPOS * DH * 2));
    float* RWt = (float*)(ws + nxt((size_t)HEADS * NPOS * RELR * 4));
    float* RHt = (float*)(ws + nxt((size_t)HEADS * NPOS * RELR * 4));
    (void)ws_size; (void)in_sizes; (void)n_in; (void)out_size;

    k_conv_fmap_t<<<(CIN * NPOS) / 256, 256, 0, stream>>>(fmap, fmap_t);
    k_conv<<<(1536 * CIN + 255) / 256, 256, 0, stream>>>(wqkv, w_bf, 1536 * CIN);
    k_conv_rel<<<(RELR * DH) / 256, 256, 0, stream>>>(relw, relw_bf);
    k_conv_rel<<<(RELR * DH) / 256, 256, 0, stream>>>(relh, relh_bf);

    k_qkv<<<dim3(24, 256), 128, 0, stream>>>(w_bf, fmap_t, q_bf, k_bf, vt_bf);

    k_rel<<<dim3(2, 256, HEADS), 128, 0, stream>>>(q_bf, relw_bf, RWt);
    k_rel<<<dim3(2, 256, HEADS), 128, 0, stream>>>(q_bf, relh_bf, RHt);

    k_attn<<<dim3(256, HEADS), 32, 0, stream>>>(q_bf, k_bf, vt_bf, RWt, RHt, out);
}
